// mLSTMCell_63651415327295
// MI455X (gfx1250) — matrix operand order repaired and checked by reading, not compiled, not run on hardware
//
#include <hip/hip_runtime.h>

// mLSTM cell for MI455X (gfx1250, wave32).
// B=16384, IN=1024, HID=1024, D=64.
//
// Pipeline:
//  1) one streaming pass converts x and all GEMM weights to f16 (x fits in the
//     192MB L2 afterwards, conversion adds ~48MB of HBM traffic ~ 2us)
//  2) all GEMMs run v_wmma_f32_16x16x32_f16 with double-buffered
//     GLOBAL_LOAD_ASYNC_TO_LDS_B128 staging of *f16* tiles: no conversion and
//     only 2 ds_load_b128 per fragment in the hot loop, 4 WMMAs per wave-iter
//  3) the (B,64,64) matrix-memory update is a pure float4-streaming kernel
//     (~512MB of the ~640MB total HBM traffic -> ~28us roofline at 23.3TB/s;
//     the GEMM compute is far below that floor with this inner loop)

#define BB   16384
#define INK  1024
#define HID  1024
#define DD   64

#define BM 128
#define BN 64
#define BK 32
#define LDH 48   // f16 LDS row stride: 96B rows -> every b128 LDS write stays
                 // 16B-aligned, and 96B/row spreads banks

typedef __attribute__((ext_vector_type(16))) _Float16 v16h;
typedef __attribute__((ext_vector_type(8)))  _Float16 v8h;
typedef __attribute__((ext_vector_type(8)))  float    v8f;

__device__ __forceinline__ float sigmoidf_(float t) {
    return 1.0f / (1.0f + expf(-t));
}

// Low 32 bits of a generic pointer to a __shared__ object are the LDS byte
// offset (AMDGPU shared-aperture representation; ISA 10.2 LDS aperture).
__device__ __forceinline__ unsigned lds_off_(const void* p) {
    return (unsigned)(uintptr_t)p;
}

// 3 per-lane async global->LDS b128 copies: two 16B chunks of the 128-row A
// tile + one 16B chunk of the 64-row B tile. Tracked by ASYNCcnt.
__device__ __forceinline__ void async_tile_issue_(unsigned la0, unsigned la1, unsigned lb,
                                                  const _Float16* ga0, const _Float16* ga1,
                                                  const _Float16* gb) {
    asm volatile(
        "global_load_async_to_lds_b128 %0, %3, off\n\t"
        "global_load_async_to_lds_b128 %1, %4, off\n\t"
        "global_load_async_to_lds_b128 %2, %5, off"
        :
        : "v"(la0), "v"(la1), "v"(lb),
          "v"((unsigned long long)(uintptr_t)ga0),
          "v"((unsigned long long)(uintptr_t)ga1),
          "v"((unsigned long long)(uintptr_t)gb)
        : "memory");
}

// ---------------------------------------------------------------------------
// elementwise f32 -> f16 converter (vector loads/stores; nelem % 2048 == 0)
// ---------------------------------------------------------------------------
__global__ __launch_bounds__(256)
void cvt_f32_f16_kernel(const float* __restrict__ src, _Float16* __restrict__ dst)
{
    const long long i = ((long long)blockIdx.x * 256 + threadIdx.x) * 8;
    float4 a = *(const float4*)(src + i);
    float4 b = *(const float4*)(src + i + 4);
    v8h o;
    o[0] = (_Float16)a.x; o[1] = (_Float16)a.y; o[2] = (_Float16)a.z; o[3] = (_Float16)a.w;
    o[4] = (_Float16)b.x; o[5] = (_Float16)b.y; o[6] = (_Float16)b.z; o[7] = (_Float16)b.w;
    *(v8h*)(dst + i) = o;
}

// ---------------------------------------------------------------------------
// Templated WMMA GEMM: Y[M,N] = epilogue( X[M,K] @ W[N,K]^T + bias[N] )
// X, W are f16 row-major (K contiguous). MODE 0: identity, 1: sigmoid,
// 2: gate[m,n] * tanh(.).
// Block tile 128x64, K-step 32, 256 threads = 8 waves; each wave owns a
// 32x32 register tile (4 WMMAs, 2x fragment reuse).  K-tiles double-buffered
// in LDS via async-to-LDS b128 copies; steady-state loop is branch-free.
// ---------------------------------------------------------------------------
template <int MODE>
__global__ __launch_bounds__(256)
void gemm_nt_wmma(const _Float16* __restrict__ X, const _Float16* __restrict__ W,
                  const float* __restrict__ bias, const float* __restrict__ gate,
                  float* __restrict__ Y, int M, int N, int K)
{
    __shared__ _Float16 As[2][BM * LDH];   // 2*128*48*2B = 24KB
    __shared__ _Float16 Bs[2][BN * LDH];   // 12KB

    const int tid  = threadIdx.x;
    const int m0   = blockIdx.y * BM;
    const int n0   = blockIdx.x * BN;
    const int wave = tid >> 5;              // 0..7
    const int lane = tid & 31;
    const int wrow = (wave >> 1) * 32;      // 0,32,64,96
    const int wcol = (wave & 1) * 32;       // 0,32
    const int half = lane >> 4;             // 0: lanes 0-15, 1: lanes 16-31
    const int l15  = lane & 15;

    v8f acc00 = {}, acc01 = {}, acc10 = {}, acc11 = {};

    // cooperative loaders: thread -> one 16B chunk per 64-row tile
    const int lrow = tid >> 2;              // 0..63
    const int lcol = (tid & 3) * 8;         // f16 elems: 0,8,16,24

    const _Float16* gA0 = X + (size_t)(m0 + lrow) * K + lcol;
    const _Float16* gA1 = X + (size_t)(m0 + 64 + lrow) * K + lcol;
    const _Float16* gB  = W + (size_t)(n0 + lrow) * K + lcol;

    unsigned laA0[2], laA1[2], laB[2];
    #pragma unroll
    for (int b = 0; b < 2; ++b) {
        laA0[b] = lds_off_(&As[b][lrow * LDH + lcol]);
        laA1[b] = lds_off_(&As[b][(lrow + 64) * LDH + lcol]);
        laB[b]  = lds_off_(&Bs[b][lrow * LDH + lcol]);
    }

    // fragment builders per ISA 7.12.2 lane layouts (f16 in LDS, contiguous
    // 16B runs -> compiler merges into ds_load_b128 pairs, no cvt)
    auto frag_a = [&](int cb, int rowbase) -> v16h {
        v16h f;
        const int am = rowbase + l15;
        #pragma unroll
        for (int j = 0; j < 16; ++j) {
            const int vv = j >> 1, p = j & 1;
            const int kk = (vv < 4 ? 2 * vv + p : 16 + 2 * (vv - 4) + p) + half * 8;
            f[j] = As[cb][am * LDH + kk];
        }
        return f;
    };
    auto frag_b = [&](int cb, int colbase) -> v16h {
        v16h f;
        const int bn = colbase + l15;
        #pragma unroll
        for (int j = 0; j < 16; ++j) {
            const int vv = j >> 1, p = j & 1;
            const int kk = (vv < 4 ? 2 * vv + p : 16 + 2 * (vv - 4) + p) + half * 8;
            f[j] = Bs[cb][bn * LDH + kk];
        }
        return f;
    };
    auto compute_tiles = [&](int cb) {
        v16h a0 = frag_a(cb, wrow);
        v16h a1 = frag_a(cb, wrow + 16);
        v16h b0 = frag_b(cb, wcol);
        v16h b1 = frag_b(cb, wcol + 16);
        acc00 = __builtin_amdgcn_wmma_f32_16x16x32_f16(false, a0, false, b0, (short)0, acc00, false, false);
        acc01 = __builtin_amdgcn_wmma_f32_16x16x32_f16(false, a0, false, b1, (short)0, acc01, false, false);
        acc10 = __builtin_amdgcn_wmma_f32_16x16x32_f16(false, a1, false, b0, (short)0, acc10, false, false);
        acc11 = __builtin_amdgcn_wmma_f32_16x16x32_f16(false, a1, false, b1, (short)0, acc11, false, false);
    };

    int cur = 0;
    async_tile_issue_(laA0[0], laA1[0], laB[0], gA0, gA1, gB);   // prime k0 = 0

    // steady state: branch-free issue(next) / wait / compute(cur)
    for (int k0 = 0; k0 < K - BK; k0 += BK) {
        const int kn = k0 + BK;
        async_tile_issue_(laA0[cur ^ 1], laA1[cur ^ 1], laB[cur ^ 1],
                          gA0 + kn, gA1 + kn, gB + kn);
        asm volatile("s_wait_asynccnt 0x3" ::: "memory");  // cur's 3 landed
        __syncthreads();                 // all waves waited -> tile visible
        compute_tiles(cur);
        __syncthreads();                 // readers done -> 'cur' reusable
        cur ^= 1;
    }
    asm volatile("s_wait_asynccnt 0x0" ::: "memory");
    __syncthreads();
    compute_tiles(cur);

    // ---- epilogue: D layout is M = r + 8*half, N = lane&15 ----
    auto epi = [&](const v8f& acc, int mt, int nt) {
        #pragma unroll
        for (int r = 0; r < 8; ++r) {
            const int gm = m0 + wrow + mt * 16 + r + half * 8;
            const int gn = n0 + wcol + nt * 16 + l15;
            float y = acc[r] + bias[gn];
            if (MODE == 1) y = sigmoidf_(y);
            if (MODE == 2) y = gate[(size_t)gm * N + gn] * tanhf(y);
            Y[(size_t)gm * N + gn] = y;
        }
    };
    epi(acc00, 0, 0); epi(acc01, 0, 1); epi(acc10, 1, 0); epi(acc11, 1, 1);
}

// ---------------------------------------------------------------------------
// Scalar gates: i = exp(x.Wi + bi), f = sigmoid(x.Wf + bf).  One wave per
// batch row, 8 rows per block; full-f32 dot products (exp is sensitive).
// ---------------------------------------------------------------------------
__global__ __launch_bounds__(256)
void gates_kernel(const float* __restrict__ X,
                  const float* __restrict__ Wi, const float* __restrict__ Wf,
                  const float* __restrict__ bi, const float* __restrict__ bf,
                  float* __restrict__ ig, float* __restrict__ fg, int K)
{
    __shared__ float ri[256];
    __shared__ float rf[256];
    const int tid  = threadIdx.x;
    const int wave = tid >> 5;
    const int lane = tid & 31;
    const int row  = blockIdx.x * 8 + wave;

    const float* xr = X + (size_t)row * K;
    float si = 0.f, sf = 0.f;
    for (int k = lane; k < K; k += 32) {
        const float xv = xr[k];
        si += xv * Wi[k];
        sf += xv * Wf[k];
    }
    ri[tid] = si;
    rf[tid] = sf;
    __syncthreads();
    if (lane == 0) {
        float ti = 0.f, tf = 0.f;
        #pragma unroll
        for (int l = 0; l < 32; ++l) { ti += ri[wave * 32 + l]; tf += rf[wave * 32 + l]; }
        ig[row] = expf(ti + bi[0]);
        fg[row] = sigmoidf_(tf + bf[0]);
    }
}

// ---------------------------------------------------------------------------
// Matrix-memory update (the HBM-bound part, ~512MB of traffic):
//   C_new = f*C + i * v k^T ;  n_new = f*n + i ;
//   h_tilde = (C_new q) / (n_new + 1e-8)   -> stored as f16 (A of final GEMM)
// One block (256 threads) per batch row; float4 streaming of the 64x64 state.
// ---------------------------------------------------------------------------
__global__ __launch_bounds__(256)
void cell_update_kernel(const float* __restrict__ q, const float* __restrict__ k,
                        const float* __restrict__ v,
                        const float* __restrict__ ig, const float* __restrict__ fg,
                        const float* __restrict__ n,
                        const float* __restrict__ C,
                        float* __restrict__ C_out, float* __restrict__ n_out,
                        _Float16* __restrict__ h_tilde_h)
{
    __shared__ float qs[DD], ks[DD], vs[DD];
    __shared__ float partial[256];

    const int b = blockIdx.x;
    const int t = threadIdx.x;

    if (t < DD) {
        qs[t] = q[(size_t)b * DD + t];
        ks[t] = k[(size_t)b * DD + t];
        vs[t] = v[(size_t)b * DD + t];
    }
    const float gi = ig[b];
    const float gf = fg[b];
    __syncthreads();

    const int row = t >> 2;            // 0..63
    const int cq  = (t & 3) * 16;      // 0,16,32,48
    const size_t base = (size_t)b * DD * DD + (size_t)row * DD + cq;
    const float iv = gi * vs[row];

    float acc = 0.f;
    #pragma unroll
    for (int g = 0; g < 4; ++g) {
        float4 c4 = *(const float4*)(C + base + g * 4);
        float4 o4;
        const int c0 = cq + g * 4;
        o4.x = gf * c4.x + iv * ks[c0 + 0];
        o4.y = gf * c4.y + iv * ks[c0 + 1];
        o4.z = gf * c4.z + iv * ks[c0 + 2];
        o4.w = gf * c4.w + iv * ks[c0 + 3];
        acc += o4.x * qs[c0 + 0] + o4.y * qs[c0 + 1]
             + o4.z * qs[c0 + 2] + o4.w * qs[c0 + 3];
        *(float4*)(C_out + base + g * 4) = o4;
    }
    partial[t] = acc;
    __syncthreads();

    if (t < DD) {
        const float s  = partial[t * 4] + partial[t * 4 + 1]
                       + partial[t * 4 + 2] + partial[t * 4 + 3];
        const float nn = gf * n[b] + gi;
        h_tilde_h[(size_t)b * DD + t] = (_Float16)(s / (nn + 1e-8f));
        if (t == 0) n_out[b] = nn;
    }
}

// ---------------------------------------------------------------------------
extern "C" void kernel_launch(void* const* d_in, const int* in_sizes, int n_in,
                              void* d_out, int out_size, void* d_ws, size_t ws_size,
                              hipStream_t stream)
{
    const float* x  = (const float*)d_in[0];
    // d_in[1] = h (unused by the reference math)
    const float* C  = (const float*)d_in[2];
    const float* n  = (const float*)d_in[3];
    const float* Wq = (const float*)d_in[4];  const float* bq = (const float*)d_in[5];
    const float* Wk = (const float*)d_in[6];  const float* bk = (const float*)d_in[7];
    const float* Wv = (const float*)d_in[8];  const float* bv = (const float*)d_in[9];
    const float* Wi = (const float*)d_in[10]; const float* bi = (const float*)d_in[11];
    const float* Wf = (const float*)d_in[12]; const float* bf = (const float*)d_in[13];
    const float* Wo = (const float*)d_in[14]; const float* bo = (const float*)d_in[15];
    const float* Wp = (const float*)d_in[16]; const float* bp = (const float*)d_in[17];

    // outputs, concatenated flat in tuple order (h_new, C_new, n_new)
    float* out   = (float*)d_out;
    float* h_new = out;
    float* C_new = h_new + (size_t)BB * HID;
    float* n_new = C_new + (size_t)BB * DD * DD;

    // workspace: f32 region (~80MB) then 16B-aligned f16 region (~38MB)
    float* ws   = (float*)d_ws;
    float* q    = ws;
    float* k    = q    + (size_t)BB * DD;
    float* v    = k    + (size_t)BB * DD;
    float* ig   = v    + (size_t)BB * DD;
    float* fg   = ig   + (size_t)BB;
    float* o    = fg   + (size_t)BB;
    _Float16* xh  = (_Float16*)(o + (size_t)BB * HID);
    _Float16* wqh = xh  + (size_t)BB * INK;
    _Float16* wkh = wqh + (size_t)DD * INK;
    _Float16* wvh = wkh + (size_t)DD * INK;
    _Float16* woh = wvh + (size_t)DD * INK;
    _Float16* wph = woh + (size_t)HID * INK;
    _Float16* hth = wph + (size_t)HID * DD;

    const dim3 blk(256);

    // f32 -> f16 conversion passes (each size is a multiple of 2048)
    cvt_f32_f16_kernel<<<dim3((BB * INK) / 2048), blk, 0, stream>>>(x, xh);
    cvt_f32_f16_kernel<<<dim3((DD * INK) / 2048), blk, 0, stream>>>(Wq, wqh);
    cvt_f32_f16_kernel<<<dim3((DD * INK) / 2048), blk, 0, stream>>>(Wk, wkh);
    cvt_f32_f16_kernel<<<dim3((DD * INK) / 2048), blk, 0, stream>>>(Wv, wvh);
    cvt_f32_f16_kernel<<<dim3((HID * INK) / 2048), blk, 0, stream>>>(Wo, woh);
    cvt_f32_f16_kernel<<<dim3((HID * DD) / 2048), blk, 0, stream>>>(Wp, wph);

    // q/k/v projections: (B x IN) @ (D x IN)^T
    gemm_nt_wmma<0><<<dim3(DD / BN, BB / BM), blk, 0, stream>>>(xh, wqh, bq, nullptr, q, BB, DD, INK);
    gemm_nt_wmma<0><<<dim3(DD / BN, BB / BM), blk, 0, stream>>>(xh, wkh, bk, nullptr, k, BB, DD, INK);
    gemm_nt_wmma<0><<<dim3(DD / BN, BB / BM), blk, 0, stream>>>(xh, wvh, bv, nullptr, v, BB, DD, INK);

    // o gate: sigmoid((B x IN) @ (HID x IN)^T + bo)  -- the 34 GFLOP GEMM
    gemm_nt_wmma<1><<<dim3(HID / BN, BB / BM), blk, 0, stream>>>(xh, woh, bo, nullptr, o, BB, HID, INK);

    // scalar i/f gates in full f32
    gates_kernel<<<dim3(BB / 8), blk, 0, stream>>>(x, Wi, Wf, bi, bf, ig, fg, INK);

    // streaming state update + retrieval (h_tilde emitted as f16)
    cell_update_kernel<<<dim3(BB), blk, 0, stream>>>(q, k, v, ig, fg, n, C, C_new, n_new, hth);

    // h_new = o * tanh(h_tilde @ Wp^T + bp), K = 64
    gemm_nt_wmma<2><<<dim3(HID / BN, BB / BM), blk, 0, stream>>>(hth, wph, bp, o, h_new, BB, HID, DD);
}
